// ColorFlowLayer_47141561040938
// MI455X (gfx1250) — matrix operand order repaired and checked by reading, not compiled, not run on hardware
//
#include <hip/hip_runtime.h>
#include <hip/hip_bf16.h>

typedef __bf16 bf16;
typedef __attribute__((ext_vector_type(16))) __bf16 v16bf;
typedef __attribute__((ext_vector_type(8)))  __bf16 bf16x8;
typedef __attribute__((ext_vector_type(8)))  float  f32x8;
typedef __attribute__((ext_vector_type(4)))  float  f32x4;

#define NNODES 50000
#define NEDGES 600000
#define NTILES (NNODES/16)    // 3125 exactly
#define NSTRIPS ((NNODES + 31) / 32)  // 1563, last strip has 16 rows
#define ETILES (NEDGES/16)    // 37500 exactly

__device__ __forceinline__ float silu_f(float x) {
  return x / (1.0f + __expf(-x));
}

// Load a 16-element bf16 fragment from two 8-element (16B) chunks.
__device__ __forceinline__ v16bf frag16(const bf16* p0, const bf16* p1) {
  bf16x8 lo = *(const bf16x8*)p0;
  bf16x8 hi = *(const bf16x8*)p1;
  v16bf r;
#pragma unroll
  for (int i = 0; i < 8; i++) { r[i] = lo[i]; r[i + 8] = hi[i]; }
  return r;
}

// ---------------------------------------------------------------------------
// Weight packing: store W[K,128] (row-major f32) as bf16 in WMMA B-fragment
// order: packed[(kblk*NT + nblk)*32 + lane][e] =
//   W[kblk*32 + (e>>3)*16 + (lane>>4)*8 + (e&7)][nblk*16 + (lane&15)]
// so a B fragment is one contiguous 32B load per lane.
// ---------------------------------------------------------------------------
__global__ void pack128_kernel(const float* __restrict__ src, int rowoff,
                               bf16* __restrict__ dst) {
  int tid = blockIdx.x * 256 + threadIdx.x;
  if (tid >= 4 * 8 * 32 * 16) return;
  int e    = tid & 15;
  int lane = (tid >> 4) & 31;
  int nblk = (tid >> 9) & 7;
  int kblk = tid >> 12;
  int r = kblk * 32 + (e >> 3) * 16 + (lane >> 4) * 8 + (e & 7);
  int c = nblk * 16 + (lane & 15);
  dst[((size_t)(kblk * 8 + nblk) * 32 + lane) * 16 + e] =
      (bf16)src[(size_t)(rowoff + r) * 128 + c];
}

// Wcat[160,384]: cols 0:128 = eW1 "src" rows, 128:256 = eW1 "dst" rows,
// 256:384 = nW1 "h/role/col" rows; K rows 0:128=h, 128:136=role, 136:144=col,
// 144:160 = zero padding.
__global__ void pack_wcat_kernel(const float* __restrict__ eW1,
                                 const float* __restrict__ nW1,
                                 bf16* __restrict__ dst) {
  int tid = blockIdx.x * 256 + threadIdx.x;
  if (tid >= 5 * 24 * 32 * 16) return;
  int e    = tid & 15;
  int lane = (tid >> 4) & 31;
  int nblk = (tid >> 9) % 24;
  int kblk = tid / (24 * 512);
  int r = kblk * 32 + (e >> 3) * 16 + (lane >> 4) * 8 + (e & 7);
  int c = nblk * 16 + (lane & 15);
  int sect = c >> 7;     // 0 = W_s, 1 = W_d, 2 = W_q
  int cc = c & 127;
  float v = 0.0f;
  if (r < 128) {
    v = (sect == 0) ? eW1[(size_t)r * 128 + cc]
      : (sect == 1) ? eW1[(size_t)(128 + r) * 128 + cc]
                    : nW1[(size_t)r * 128 + cc];
  } else if (r < 136) {
    int rr = r - 128;
    v = (sect == 0) ? eW1[(size_t)(272 + rr) * 128 + cc]
      : (sect == 1) ? eW1[(size_t)(280 + rr) * 128 + cc]
                    : nW1[(size_t)(256 + rr) * 128 + cc];
  } else if (r < 144) {
    int rr = r - 136;
    v = (sect == 0) ? eW1[(size_t)(288 + rr) * 128 + cc]
      : (sect == 1) ? eW1[(size_t)(296 + rr) * 128 + cc]
                    : nW1[(size_t)(264 + rr) * 128 + cc];
  }
  dst[((size_t)(kblk * 24 + nblk) * 32 + lane) * 16 + e] = (bf16)v;
}

// X[N,160] bf16 = [h | role_emb[role] | color_emb[col] | 0-pad]
__global__ void build_xn_kernel(const float* __restrict__ h,
                                const int* __restrict__ role,
                                const int* __restrict__ color,
                                const float* __restrict__ role_emb,
                                const float* __restrict__ color_emb,
                                bf16* __restrict__ Xn) {
  int tid = blockIdx.x * 256 + threadIdx.x;
  if (tid >= NNODES * 160) return;
  int c = tid % 160;
  int n = tid / 160;
  float v = 0.0f;
  if (c < 128)      v = h[(size_t)n * 128 + c];
  else if (c < 136) v = role_emb[role[n] * 8 + (c - 128)];
  else if (c < 144) v = color_emb[color[n] * 8 + (c - 136)];
  Xn[tid] = (bf16)v;
}

// pr[8][128] = rel_emb @ eW1[256:272] + eb1   (tiny; VALU)
__global__ void pr_kernel(const float* __restrict__ rel_emb,
                          const float* __restrict__ eW1,
                          const float* __restrict__ eb1,
                          float* __restrict__ prt) {
  int tid = blockIdx.x * 256 + threadIdx.x;
  if (tid >= 8 * 128) return;
  int r = tid >> 7, c = tid & 127;
  float acc = eb1[c];
#pragma unroll
  for (int k = 0; k < 16; k++)
    acc += rel_emb[r * 16 + k] * eW1[(size_t)(256 + k) * 128 + c];
  prt[tid] = acc;
}

// ---------------------------------------------------------------------------
// K1: P[N,384] = X[N,160] @ Wcat[160,384]   (ps | pd | q), f32 accum via WMMA
// 32 rows (two 16-row A-tile sets) per wave; each B fragment batched into
// registers before the WMMA chain and reused by both M-tiles.
// ---------------------------------------------------------------------------
__global__ void __launch_bounds__(128) gemm_p_kernel(
    const bf16* __restrict__ Xn, const bf16* __restrict__ Wp,
    float* __restrict__ P) {
  int strip = blockIdx.x * 4 + (threadIdx.x >> 5);
  if (strip >= NSTRIPS) return;
  int lane = threadIdx.x & 31;
  int ln = lane & 15, lh = lane >> 4;
  int row0 = strip * 32;
  bool has2 = (row0 + 16) < NNODES;   // wave-uniform

  v16bf a0[5], a1[5];
  const bf16* x0 = Xn + (size_t)(row0 + ln) * 160;
#pragma unroll
  for (int kb = 0; kb < 5; kb++) {
    const bf16* p = x0 + kb * 32 + lh * 8;
    a0[kb] = frag16(p, p + 16);
  }
  if (has2) {
    const bf16* x1 = Xn + (size_t)(row0 + 16 + ln) * 160;
#pragma unroll
    for (int kb = 0; kb < 5; kb++) {
      const bf16* p = x1 + kb * 32 + lh * 8;
      a1[kb] = frag16(p, p + 16);
    }
  }
#pragma unroll 1
  for (int nt = 0; nt < 24; nt++) {
    v16bf b[5];
#pragma unroll
    for (int kb = 0; kb < 5; kb++) {
      const bf16* bp = Wp + ((size_t)(kb * 24 + nt) * 32 + lane) * 16;
      b[kb] = frag16(bp, bp + 8);
    }
    f32x8 acc0 = {0.f, 0.f, 0.f, 0.f, 0.f, 0.f, 0.f, 0.f};
#pragma unroll
    for (int kb = 0; kb < 5; kb++)
      acc0 = __builtin_amdgcn_wmma_f32_16x16x32_bf16(
          false, a0[kb], false, b[kb], (short)0, acc0, false, false);
    float* p0 = P + (size_t)row0 * 384 + nt * 16 + ln;
#pragma unroll
    for (int v = 0; v < 8; v++)
      p0[(size_t)(v + 8 * lh) * 384] = acc0[v];
    if (has2) {
      f32x8 acc1 = {0.f, 0.f, 0.f, 0.f, 0.f, 0.f, 0.f, 0.f};
#pragma unroll
      for (int kb = 0; kb < 5; kb++)
        acc1 = __builtin_amdgcn_wmma_f32_16x16x32_bf16(
            false, a1[kb], false, b[kb], (short)0, acc1, false, false);
      float* p1 = P + (size_t)(row0 + 16) * 384 + nt * 16 + ln;
#pragma unroll
      for (int v = 0; v < 8; v++)
        p1[(size_t)(v + 8 * lh) * 384] = acc1[v];
    }
  }
}

// ---------------------------------------------------------------------------
// K2: per 16-edge tile: Y1 = silu(ps[src]+pd[dst]+pr[rel]) -> bf16 in LDS,
// M = silu(Y1 @ eW2 + eb2) via WMMA, then f32 atomics into agg[dst].
// eW2 (packed, 32KB) staged in LDS once per block; edge indices prefetched.
// 4 waves / 128-thread block; E divides 16 exactly -> EXEC full at WMMA.
// ---------------------------------------------------------------------------
__global__ void __launch_bounds__(128) edge_kernel(
    const int* __restrict__ eidx, const int* __restrict__ erel,
    const float* __restrict__ P, const float* __restrict__ prt,
    const bf16* __restrict__ W2p, const float* __restrict__ eb2,
    float* __restrict__ agg) {
  __shared__ __align__(16) bf16 W2s[16384];   // 32 KB packed eW2
  __shared__ __align__(16) bf16 Y1[4][16][128];
  __shared__ int SRCS[4][16];
  __shared__ int DSTS[4][16];
  __shared__ int RELS[4][16];
  int w    = threadIdx.x >> 5;
  int lane = threadIdx.x & 31;
  int ln = lane & 15, lh = lane >> 4;
  int ebase = (blockIdx.x * 4 + w) * 16;

  // stage packed eW2 into LDS (2048 x 16B)
  for (int i = threadIdx.x; i < 2048; i += 128)
    ((f32x4*)W2s)[i] = ((const f32x4*)W2p)[i];

  // prefetch this wave's 16 edge index triplets
  if (lane < 16) {
    int eid = ebase + lane;
    SRCS[w][lane] = eidx[eid];
    DSTS[w][lane] = eidx[NEDGES + eid];
    RELS[w][lane] = erel[eid];
  }
  __syncthreads();

  // gather + SiLU -> Y1 (bf16)
#pragma unroll 4
  for (int e = 0; e < 16; e++) {
    int src = SRCS[w][e];
    int dst = DSTS[w][e];
    int rel = RELS[w][e];
    f32x4 s = *(const f32x4*)(P + (size_t)src * 384 + lane * 4);
    f32x4 d = *(const f32x4*)(P + (size_t)dst * 384 + 128 + lane * 4);
    f32x4 r = *(const f32x4*)(prt + rel * 128 + lane * 4);
    f32x4 y = s + d + r;
    bf16* yp = &Y1[w][e][lane * 4];
#pragma unroll
    for (int j = 0; j < 4; j++) yp[j] = (bf16)silu_f(y[j]);
  }
  asm volatile("s_wait_dscnt 0" ::: "memory");  // LDS in-order per wave

  v16bf a[4];
#pragma unroll
  for (int kb = 0; kb < 4; kb++) {
    const bf16* p = &Y1[w][ln][kb * 32 + lh * 8];
    a[kb] = frag16(p, p + 16);
  }
#pragma unroll 1
  for (int nt = 0; nt < 8; nt++) {
    v16bf b[4];
#pragma unroll
    for (int kb = 0; kb < 4; kb++) {
      const bf16* bp = &W2s[((kb * 8 + nt) * 32 + lane) * 16];
      b[kb] = frag16(bp, bp + 8);
    }
    f32x8 acc = {0.f, 0.f, 0.f, 0.f, 0.f, 0.f, 0.f, 0.f};
#pragma unroll
    for (int kb = 0; kb < 4; kb++)
      acc = __builtin_amdgcn_wmma_f32_16x16x32_bf16(
          false, a[kb], false, b[kb], (short)0, acc, false, false);
    float bias = eb2[nt * 16 + ln];
#pragma unroll
    for (int v = 0; v < 8; v++) {
      float m = silu_f(acc[v] + bias);
      int dst = DSTS[w][v + 8 * lh];
      atomicAdd(agg + (size_t)dst * 128 + nt * 16 + ln, m);
    }
  }
}

// ---------------------------------------------------------------------------
// K3: per 16-node tile: y1 = silu(q + agg@nW1[128:256] + nb1),
// U = y1@nW2 + nb2, x = h + U, LayerNorm via shfl_xor over 16-lane groups.
// ---------------------------------------------------------------------------
__global__ void __launch_bounds__(128) node_kernel(
    const float* __restrict__ agg, const float* __restrict__ P,
    const bf16* __restrict__ N1p, const bf16* __restrict__ N2p,
    const float* __restrict__ h, const float* __restrict__ nb1,
    const float* __restrict__ nb2, const float* __restrict__ ln_g,
    const float* __restrict__ ln_b, float* __restrict__ out) {
  __shared__ __align__(16) bf16 T[4][16][128];
  int w    = threadIdx.x >> 5;
  int lane = threadIdx.x & 31;
  int ln = lane & 15, lh = lane >> 4;
  int tile = blockIdx.x * 4 + (threadIdx.x >> 5);
  if (tile >= NTILES) return;
  int nbase = tile * 16;

  // stage agg tile as bf16 A-matrix
#pragma unroll 4
  for (int e = 0; e < 16; e++) {
    f32x4 av = *(const f32x4*)(agg + (size_t)(nbase + e) * 128 + lane * 4);
    bf16* tp = &T[w][e][lane * 4];
#pragma unroll
    for (int j = 0; j < 4; j++) tp[j] = (bf16)av[j];
  }
  asm volatile("s_wait_dscnt 0" ::: "memory");

  v16bf a[4];
#pragma unroll
  for (int kb = 0; kb < 4; kb++) {
    const bf16* p = &T[w][ln][kb * 32 + lh * 8];
    a[kb] = frag16(p, p + 16);
  }
  f32x8 acc[8];
#pragma unroll 1
  for (int nt = 0; nt < 8; nt++) {
    v16bf b[4];
#pragma unroll
    for (int kb = 0; kb < 4; kb++) {
      const bf16* bp = N1p + ((size_t)(kb * 8 + nt) * 32 + lane) * 16;
      b[kb] = frag16(bp, bp + 8);
    }
    f32x8 c = {0.f, 0.f, 0.f, 0.f, 0.f, 0.f, 0.f, 0.f};
#pragma unroll
    for (int kb = 0; kb < 4; kb++)
      c = __builtin_amdgcn_wmma_f32_16x16x32_bf16(
          false, a[kb], false, b[kb], (short)0, c, false, false);
    acc[nt] = c;
  }
  // y1 = silu(G + q + nb1) -> back into T as bf16
#pragma unroll
  for (int nt = 0; nt < 8; nt++) {
    float b1 = nb1[nt * 16 + ln];
#pragma unroll
    for (int v = 0; v < 8; v++) {
      int m = v + 8 * lh;
      float q = P[(size_t)(nbase + m) * 384 + 256 + nt * 16 + ln];
      T[w][m][nt * 16 + ln] = (bf16)silu_f(acc[nt][v] + q + b1);
    }
  }
  asm volatile("s_wait_dscnt 0" ::: "memory");
#pragma unroll
  for (int kb = 0; kb < 4; kb++) {
    const bf16* p = &T[w][ln][kb * 32 + lh * 8];
    a[kb] = frag16(p, p + 16);
  }
#pragma unroll 1
  for (int nt = 0; nt < 8; nt++) {
    v16bf b[4];
#pragma unroll
    for (int kb = 0; kb < 4; kb++) {
      const bf16* bp = N2p + ((size_t)(kb * 8 + nt) * 32 + lane) * 16;
      b[kb] = frag16(bp, bp + 8);
    }
    f32x8 c = {0.f, 0.f, 0.f, 0.f, 0.f, 0.f, 0.f, 0.f};
#pragma unroll
    for (int kb = 0; kb < 4; kb++)
      c = __builtin_amdgcn_wmma_f32_16x16x32_bf16(
          false, a[kb], false, b[kb], (short)0, c, false, false);
    acc[nt] = c;
  }
  // x = h + U + nb2; LayerNorm
  float s1[8], s2[8];
#pragma unroll
  for (int v = 0; v < 8; v++) { s1[v] = 0.f; s2[v] = 0.f; }
#pragma unroll
  for (int nt = 0; nt < 8; nt++) {
    float b2 = nb2[nt * 16 + ln];
#pragma unroll
    for (int v = 0; v < 8; v++) {
      int m = v + 8 * lh;
      float x = acc[nt][v] + b2 + h[(size_t)(nbase + m) * 128 + nt * 16 + ln];
      acc[nt][v] = x;
      s1[v] += x;
      s2[v] += x * x;
    }
  }
#pragma unroll
  for (int v = 0; v < 8; v++) {
#pragma unroll
    for (int msk = 1; msk < 16; msk <<= 1) {
      s1[v] += __shfl_xor(s1[v], msk, 32);
      s2[v] += __shfl_xor(s2[v], msk, 32);
    }
  }
#pragma unroll
  for (int nt = 0; nt < 8; nt++) {
    int c = nt * 16 + ln;
    float g = ln_g[c], bb = ln_b[c];
#pragma unroll
    for (int v = 0; v < 8; v++) {
      float mu  = s1[v] * (1.0f / 128.0f);
      float var = s2[v] * (1.0f / 128.0f) - mu * mu;
      float inv = rsqrtf(var + 1e-5f);
      int m = v + 8 * lh;
      out[(size_t)(nbase + m) * 128 + c] = (acc[nt][v] - mu) * inv * g + bb;
    }
  }
}

extern "C" void kernel_launch(void* const* d_in, const int* in_sizes, int n_in,
                              void* d_out, int out_size, void* d_ws,
                              size_t ws_size, hipStream_t stream) {
  (void)in_sizes; (void)n_in; (void)out_size; (void)ws_size;
  const float* h        = (const float*)d_in[0];
  const int*   eidx     = (const int*)d_in[1];
  const int*   erel     = (const int*)d_in[2];
  const int*   ncol     = (const int*)d_in[3];
  const int*   nrole    = (const int*)d_in[4];
  const float* rel_emb  = (const float*)d_in[5];
  const float* role_emb = (const float*)d_in[6];
  const float* col_emb  = (const float*)d_in[7];
  const float* eW1      = (const float*)d_in[8];
  const float* eb1      = (const float*)d_in[9];
  const float* eW2      = (const float*)d_in[10];
  const float* eb2      = (const float*)d_in[11];
  const float* nW1      = (const float*)d_in[12];
  const float* nb1      = (const float*)d_in[13];
  const float* nW2      = (const float*)d_in[14];
  const float* nb2      = (const float*)d_in[15];
  const float* lng      = (const float*)d_in[16];
  const float* lnb      = (const float*)d_in[17];

  char* w = (char*)d_ws;
  float* P    = (float*)w;  w += (size_t)NNODES * 384 * 4;  // ps|pd|q
  float* agg  = (float*)w;  w += (size_t)NNODES * 128 * 4;
  bf16*  Xn   = (bf16*)w;   w += (size_t)NNODES * 160 * 2;
  bf16*  Wcp  = (bf16*)w;   w += (size_t)160 * 384 * 2;
  bf16*  W2p  = (bf16*)w;   w += (size_t)128 * 128 * 2;
  bf16*  N1p  = (bf16*)w;   w += (size_t)128 * 128 * 2;
  bf16*  N2p  = (bf16*)w;   w += (size_t)128 * 128 * 2;
  float* prt  = (float*)w;  w += (size_t)8 * 128 * 4;

  hipMemsetAsync(agg, 0, (size_t)NNODES * 128 * 4, stream);
  build_xn_kernel<<<(NNODES * 160 + 255) / 256, 256, 0, stream>>>(
      h, nrole, ncol, role_emb, col_emb, Xn);
  pack_wcat_kernel<<<(5 * 24 * 32 * 16 + 255) / 256, 256, 0, stream>>>(eW1, nW1, Wcp);
  pack128_kernel<<<64, 256, 0, stream>>>(eW2, 0, W2p);
  pack128_kernel<<<64, 256, 0, stream>>>(nW1, 128, N1p);
  pack128_kernel<<<64, 256, 0, stream>>>(nW2, 0, N2p);
  pr_kernel<<<4, 256, 0, stream>>>(rel_emb, eW1, eb1, prt);

  gemm_p_kernel<<<(NSTRIPS + 3) / 4, 128, 0, stream>>>(Xn, Wcp, P);
  edge_kernel<<<ETILES / 4, 128, 0, stream>>>(eidx, erel, P, prt, W2p, eb2, agg);
  node_kernel<<<(NTILES + 3) / 4, 128, 0, stream>>>(
      agg, P, N1p, N2p, h, nb1, nb2, lng, lnb, (float*)d_out);
}
